// DGCNNForExplainerPyG_9534827397390
// MI455X (gfx1250) — compile-verified, matrix operand order checked by reading, not run on hardware
//
#include <hip/hip_runtime.h>
#include <hip/hip_bf16.h>

#define N_NODES 50000
#define C_IN    32
#define H_HID   32
#define E_EDGES 1600000
#define FC_HIDN 64
#define NCLASS  2
#define BN_EPS  1e-5f
#define NH      (N_NODES * H_HID)        // 1,600,000
#define CHUNK   65536
#define NCHUNK  ((NH + CHUNK - 1) / CHUNK)  // 25

typedef __attribute__((ext_vector_type(16))) _Float16 v16h;
typedef __attribute__((ext_vector_type(8)))  float    v8f;

// K index held by (half, vgpr-slot e) for 16-bit 16x32 A operand (ISA 7.12.2).
// lane = 16*half + m ; element e = 2*v + j (v = VGPR 0..7, j = lo/hi half)
__device__ __forceinline__ int kmap(int half, int e) {
    int v = e >> 1, j = e & 1;
    return (v < 4) ? (half * 8 + 2 * v + j)
                   : (16 + half * 8 + 2 * (v - 4) + j);
}

// ---------------------------------------------------------------------------
// Stage 1: BatchNorm (eval) + zero the two scatter accumulators + seed fc1 acc
// ---------------------------------------------------------------------------
__global__ void bn_init_kernel(const float* __restrict__ x,
                               const float* __restrict__ gamma,
                               const float* __restrict__ beta,
                               const float* __restrict__ mean,
                               const float* __restrict__ var,
                               const float* __restrict__ fc1_b,
                               float* __restrict__ xb,
                               float* __restrict__ agg0,
                               float* __restrict__ agg,
                               float* __restrict__ acc) {
    int i = blockIdx.x * blockDim.x + threadIdx.x;
    if (i < N_NODES * C_IN) {
        int c = i & (C_IN - 1);
        float s = rsqrtf(var[c] + BN_EPS) * gamma[c];
        xb[i]   = (x[i] - mean[c]) * s + beta[c];
        agg0[i] = 0.0f;
        agg[i]  = 0.0f;
    }
    if (i < FC_HIDN) acc[i] = fc1_b[i];   // fc1 bias seeds the atomic accumulator
}

// ---------------------------------------------------------------------------
// Stage 2: pack B operands (W[0] and W[1]+W[2], each split into two 16-col
// tiles) into the per-lane WMMA layout so the GEMM wave loads them as v16h.
// Bops[op][lane][e], op = ws*2 + ntile.
// ---------------------------------------------------------------------------
__global__ void prep_w_kernel(const float* __restrict__ W,
                              _Float16* __restrict__ Bops) {
    int t = blockIdx.x * blockDim.x + threadIdx.x;
    if (t >= 4 * 32) return;
    int op = t >> 5, lane = t & 31;
    int ntile = op & 1;            // column tile: 0 -> cols 0..15, 1 -> 16..31
    int ws    = op >> 1;           // 0 -> W[0], 1 -> W[1]+W[2]
    int n = lane & 15, h = lane >> 4;
    for (int e = 0; e < 16; ++e) {
        int k = kmap(h, e);
        int col = ntile * 16 + n;
        float val;
        if (ws == 0) val = W[k * H_HID + col];
        else         val = W[1 * C_IN * H_HID + k * H_HID + col]
                         + W[2 * C_IN * H_HID + k * H_HID + col];
        Bops[(op * 32 + lane) * 16 + e] = (_Float16)val;
    }
}

// ---------------------------------------------------------------------------
// Stage 3: edge gather + scatter. 4 threads per edge, 8 channels each.
// agg[dst]  += w * xb[src]   ;   agg0[dst] += (src==dst) * xb[src]
// ---------------------------------------------------------------------------
__global__ void __launch_bounds__(256)
edge_scatter_kernel(const float* __restrict__ xb,
                    const long long* __restrict__ eidx,   // int64 [2, E]
                    const float* __restrict__ ew,
                    float* __restrict__ agg0,
                    float* __restrict__ agg) {
    long long gid = (long long)blockIdx.x * blockDim.x + threadIdx.x;
    int e = (int)(gid >> 2);
    int q = (int)(gid & 3);
    if (e >= E_EDGES) return;
    int src = (int)eidx[e];
    int dst = (int)eidx[(long long)E_EDGES + e];
    float w = ew[e];
    bool self = (src == dst);
    const float4* xr = (const float4*)(xb + (size_t)src * C_IN + q * 8);
    float4 v0 = xr[0];
    float4 v1 = xr[1];
    float* ar  = agg  + (size_t)dst * C_IN + q * 8;
    float* a0r = agg0 + (size_t)dst * C_IN + q * 8;
    atomicAdd(ar + 0, w * v0.x); atomicAdd(ar + 1, w * v0.y);
    atomicAdd(ar + 2, w * v0.z); atomicAdd(ar + 3, w * v0.w);
    atomicAdd(ar + 4, w * v1.x); atomicAdd(ar + 5, w * v1.y);
    atomicAdd(ar + 6, w * v1.z); atomicAdd(ar + 7, w * v1.w);
    if (self) {
        atomicAdd(a0r + 0, v0.x); atomicAdd(a0r + 1, v0.y);
        atomicAdd(a0r + 2, v0.z); atomicAdd(a0r + 3, v0.w);
        atomicAdd(a0r + 4, v1.x); atomicAdd(a0r + 5, v1.y);
        atomicAdd(a0r + 6, v1.z); atomicAdd(a0r + 7, v1.w);
    }
}

// ---------------------------------------------------------------------------
// Stage 4: hidden = relu(agg0 @ W0 + agg @ (W1+W2)) via WMMA.
// One wave (32 lanes) per 16-node tile; K = C = 32 -> one WMMA per (A,B) pair.
// EXEC is all-1s (block == one full wave, no divergence around WMMA).
// ---------------------------------------------------------------------------
__global__ void __launch_bounds__(32)
gemm_relu_kernel(const float* __restrict__ agg0,
                 const float* __restrict__ agg,
                 const _Float16* __restrict__ Bops,
                 float* __restrict__ hidden) {
    int tile = blockIdx.x;           // 3125 tiles of 16 nodes
    int lane = threadIdx.x;
    int m = lane & 15, h = lane >> 4;

    const float* a0row = agg0 + ((size_t)tile * 16 + m) * C_IN;
    const float* a1row = agg  + ((size_t)tile * 16 + m) * C_IN;
    v16h a0, a1;
#pragma unroll
    for (int e = 0; e < 16; ++e) {
        int k = kmap(h, e);
        a0[e] = (_Float16)a0row[k];
        a1[e] = (_Float16)a1row[k];
    }
    const v16h* B = (const v16h*)Bops;          // [4][32] lanes, 32B each
    v16h bw0n0 = B[0 * 32 + lane];
    v16h bw0n1 = B[1 * 32 + lane];
    v16h bwsn0 = B[2 * 32 + lane];
    v16h bwsn1 = B[3 * 32 + lane];

    v8f c0 = {}, c1 = {};
    c0 = __builtin_amdgcn_wmma_f32_16x16x32_f16(false, a0, false, bw0n0, (short)0, c0, false, false);
    c0 = __builtin_amdgcn_wmma_f32_16x16x32_f16(false, a1, false, bwsn0, (short)0, c0, false, false);
    c1 = __builtin_amdgcn_wmma_f32_16x16x32_f16(false, a0, false, bw0n1, (short)0, c1, false, false);
    c1 = __builtin_amdgcn_wmma_f32_16x16x32_f16(false, a1, false, bwsn1, (short)0, c1, false, false);

    int n = lane & 15;
#pragma unroll
    for (int r = 0; r < 8; ++r) {               // C/D: VGPR r -> row r + 8*half
        int row = tile * 16 + r + 8 * h;
        float v0 = c0[r] > 0.0f ? c0[r] : 0.0f;
        float v1 = c1[r] > 0.0f ? c1[r] : 0.0f;
        hidden[(size_t)row * H_HID + n]      = v0;
        hidden[(size_t)row * H_HID + 16 + n] = v1;
    }
}

// ---------------------------------------------------------------------------
// Stage 5: fc1 matvec acc[j] += dot(hidden, fc1_w[j, chunk]). fc1_w (410 MB)
// streamed exactly once; hidden (6.4 MB) stays resident in the 192 MB L2.
// ---------------------------------------------------------------------------
__global__ void __launch_bounds__(256)
fc1_kernel(const float* __restrict__ hidden,
           const float* __restrict__ fc1_w,
           float* __restrict__ acc) {
    __shared__ float red[256];
    int j     = blockIdx.x & 63;
    int chunk = blockIdx.x >> 6;
    int base  = chunk * CHUNK;
    int limit = min(base + CHUNK, NH);
    const float4* h4 = (const float4*)hidden;
    const float4* w4 = (const float4*)(fc1_w + (size_t)j * NH);
    float s = 0.0f;
    for (int i = base + threadIdx.x * 4; i < limit; i += 256 * 4) {
        __builtin_prefetch(&w4[(i >> 2) + 2048], 0, 0);   // global_prefetch_b8
        float4 hv = h4[i >> 2];
        float4 wv = w4[i >> 2];
        s += hv.x * wv.x + hv.y * wv.y + hv.z * wv.z + hv.w * wv.w;
    }
    red[threadIdx.x] = s;
    __syncthreads();
    for (int off = 128; off > 0; off >>= 1) {
        if (threadIdx.x < off) red[threadIdx.x] += red[threadIdx.x + off];
        __syncthreads();
    }
    if (threadIdx.x == 0) atomicAdd(&acc[j], red[0]);
}

// ---------------------------------------------------------------------------
// Stage 6: out = relu(acc) @ fc2_w.T + fc2_b
// ---------------------------------------------------------------------------
__global__ void head_kernel(const float* __restrict__ acc,
                            const float* __restrict__ fc2_w,
                            const float* __restrict__ fc2_b,
                            float* __restrict__ out) {
    __shared__ float hsm[FC_HIDN];
    int t = threadIdx.x;
    if (t < FC_HIDN) hsm[t] = acc[t] > 0.0f ? acc[t] : 0.0f;
    __syncthreads();
    if (t < NCLASS) {
        float s = fc2_b[t];
        for (int jj = 0; jj < FC_HIDN; ++jj) s += hsm[jj] * fc2_w[t * FC_HIDN + jj];
        out[t] = s;
    }
}

// ---------------------------------------------------------------------------
extern "C" void kernel_launch(void* const* d_in, const int* in_sizes, int n_in,
                              void* d_out, int out_size, void* d_ws, size_t ws_size,
                              hipStream_t stream) {
    const float*     x      = (const float*)d_in[0];
    const float*     ew     = (const float*)d_in[1];
    const float*     W      = (const float*)d_in[2];
    const float*     gamma  = (const float*)d_in[3];
    const float*     beta   = (const float*)d_in[4];
    const float*     mean   = (const float*)d_in[5];
    const float*     var    = (const float*)d_in[6];
    const float*     fc1_w  = (const float*)d_in[7];
    const float*     fc1_b  = (const float*)d_in[8];
    const float*     fc2_w  = (const float*)d_in[9];
    const float*     fc2_b  = (const float*)d_in[10];
    const long long* eidx   = (const long long*)d_in[11];   // int64 [2, E]

    char* ws = (char*)d_ws;
    float*    xb     = (float*)(ws + 0);                    // 6.4 MB
    float*    agg0   = (float*)(ws + 6400000);              // 6.4 MB
    float*    agg    = (float*)(ws + 12800000);             // 6.4 MB
    float*    hidden = (float*)(ws + 19200000);             // 6.4 MB
    _Float16* Bops   = (_Float16*)(ws + 25600000);          // 4 KB
    float*    acc    = (float*)(ws + 25604096);             // 256 B
    float*    out    = (float*)d_out;

    const int nc = N_NODES * C_IN;
    hipLaunchKernelGGL(bn_init_kernel, dim3((nc + 255) / 256), dim3(256), 0, stream,
                       x, gamma, beta, mean, var, fc1_b, xb, agg0, agg, acc);
    hipLaunchKernelGGL(prep_w_kernel, dim3(1), dim3(128), 0, stream, W, Bops);
    long long tot = (long long)E_EDGES * 4;
    hipLaunchKernelGGL(edge_scatter_kernel, dim3((unsigned)((tot + 255) / 256)), dim3(256),
                       0, stream, xb, eidx, ew, agg0, agg);
    hipLaunchKernelGGL(gemm_relu_kernel, dim3(N_NODES / 16), dim3(32), 0, stream,
                       agg0, agg, Bops, hidden);
    hipLaunchKernelGGL(fc1_kernel, dim3(64 * NCHUNK), dim3(256), 0, stream,
                       hidden, fc1_w, acc);
    hipLaunchKernelGGL(head_kernel, dim3(1), dim3(64), 0, stream,
                       acc, fc2_w, fc2_b, out);
}
